// GPT_67095979098978
// MI455X (gfx1250) — compile-verified, matrix operand order checked by reading, not run on hardware
//
#include <hip/hip_runtime.h>

// GPT-2 small forward on gfx1250 (MI455X), bf16 WMMA everywhere.
// Sizes from the reference.
static constexpr int Bq = 2, Sq = 1024, Dq = 768, Lq = 12, Hq = 12, Fq = 3072, Vq = 50257;
static constexpr int Mq = Bq * Sq; // 2048 rows in the token dimension

typedef __attribute__((ext_vector_type(16))) __bf16 v16bf;
typedef __attribute__((ext_vector_type(8)))  __bf16 v8bf;
typedef __attribute__((ext_vector_type(8)))  float  v8f;

// ---- helpers -------------------------------------------------------------

__device__ __forceinline__ __bf16 f2bf(float f) {
  // round-to-nearest-even fp32 -> bf16
  unsigned int u = __float_as_uint(f);
  unsigned int r = (u + 0x7FFFu + ((u >> 16) & 1u)) >> 16;
  unsigned short s = (unsigned short)r;
  return __builtin_bit_cast(__bf16, s);
}

// Per-lane 16-element bf16 fragment: two contiguous 16-byte chunks.
// Matches the CDNA5 16-bit A-matrix layout: elems 0..7 -> K = 8*half + 0..7,
// elems 8..15 -> K = 16 + 8*half + 0..7 (caller bakes 8*half into the pointer).
__device__ __forceinline__ v16bf load_frag(const __bf16* p) {
  v8bf lo = *reinterpret_cast<const v8bf*>(p);
  v8bf hi = *reinterpret_cast<const v8bf*>(p + 16);
  return __builtin_shufflevector(lo, hi, 0,1,2,3,4,5,6,7,8,9,10,11,12,13,14,15);
}

__device__ __forceinline__ float gelu_f(float x) {
  return 0.5f * x * (1.0f + erff(x * 0.70710678118654752f));
}

// ---- data movement / prep kernels ---------------------------------------

__global__ __launch_bounds__(256) void cvt_bf16_kernel(const float* __restrict__ in,
                                                       __bf16* __restrict__ out,
                                                       long long n) {
  long long i = (long long)blockIdx.x * 256 + threadIdx.x;
  if (i < n) out[i] = f2bf(in[i]);
}

// fp32 [K,N] row-major -> bf16 [N,K] row-major (weight transpose for B frags)
__global__ __launch_bounds__(256) void transpose_bf16_kernel(const float* __restrict__ in,
                                                             __bf16* __restrict__ out,
                                                             int K_, int N_) {
  long long i = (long long)blockIdx.x * 256 + threadIdx.x;
  long long total = (long long)K_ * N_;
  if (i >= total) return;
  int n = (int)(i % N_);
  int k = (int)(i / N_);
  out[(size_t)n * K_ + k] = f2bf(in[i]);
}

__global__ __launch_bounds__(256) void embed_kernel(const int* __restrict__ ids,
                                                    const float* __restrict__ wte,
                                                    const float* __restrict__ wpe,
                                                    float* __restrict__ x) {
  int i = blockIdx.x * 256 + threadIdx.x;
  if (i >= Mq * Dq) return;
  int d = i % Dq, row = i / Dq, s = row % Sq;
  int tok = ids[row];
  x[i] = wte[(size_t)tok * Dq + d] + wpe[(size_t)s * Dq + d];
}

// LayerNorm: one block per row, fp32 in -> bf16 out
__global__ __launch_bounds__(256) void ln_kernel(const float* __restrict__ x,
                                                 const float* __restrict__ sc,
                                                 const float* __restrict__ bi,
                                                 __bf16* __restrict__ out) {
  int row = blockIdx.x;
  int tid = threadIdx.x;
  const float* xr = x + (size_t)row * Dq;
  float s = 0.f, q = 0.f;
  for (int i = tid; i < Dq; i += 256) { float v = xr[i]; s += v; q += v * v; }
  __shared__ float ss[256], sq[256];
  ss[tid] = s; sq[tid] = q; __syncthreads();
  for (int st = 128; st > 0; st >>= 1) {
    if (tid < st) { ss[tid] += ss[tid + st]; sq[tid] += sq[tid + st]; }
    __syncthreads();
  }
  float mu  = ss[0] * (1.0f / Dq);
  float var = fmaxf(sq[0] * (1.0f / Dq) - mu * mu, 0.f);
  float rs  = rsqrtf(var + 1e-5f);
  for (int i = tid; i < Dq; i += 256)
    out[(size_t)row * Dq + i] = f2bf((xr[i] - mu) * rs * sc[i] + bi[i]);
}

// qkv fp32 [M, 3D] -> Qb bf16 [B*H,S,64] (scaled by 1/8), Kb bf16 [B*H,S,64],
// VT bf16 [B*H,64,S] (transposed per head for P@V B-fragments)
__global__ __launch_bounds__(256) void qkv_split_kernel(const float* __restrict__ qkv,
                                                        __bf16* __restrict__ Qb,
                                                        __bf16* __restrict__ Kb,
                                                        __bf16* __restrict__ VT) {
  int i = blockIdx.x * 256 + threadIdx.x;
  if (i >= Mq * Dq) return;
  int d = i % Dq, row = i / Dq;
  int b = row / Sq, s = row % Sq;
  int hh = d >> 6, dk = d & 63;
  size_t base = (size_t)row * (3 * Dq);
  size_t bh = (size_t)b * Hq + hh;
  Qb[(bh * Sq + s) * 64 + dk] = f2bf(qkv[base + d] * 0.125f);
  Kb[(bh * Sq + s) * 64 + dk] = f2bf(qkv[base + Dq + d]);
  VT[(bh * 64 + dk) * Sq + s] = f2bf(qkv[base + 2 * Dq + d]);
}

// ---- WMMA GEMM: out[M,N] = A[M,K](bf16) @ BT[N,K]^T(bf16) (+bias)(+resid)(gelu) ----
// 256 threads = 8 waves; block tile 128x128; each wave 32x64 (2x4 of 16x16).
template <bool GELU, bool OUTBF16>
__global__ __launch_bounds__(256) void gemm_kernel(const __bf16* __restrict__ A,
                                                   const __bf16* __restrict__ BT,
                                                   const float* __restrict__ bias,
                                                   const float* __restrict__ resid,
                                                   void* __restrict__ outv,
                                                   int N, int K) {
  const int lane = threadIdx.x & 31;
  const int wave = threadIdx.x >> 5;
  const int wm = wave >> 1;          // 0..3
  const int wn = wave & 1;           // 0..1
  const int lm = lane & 15;
  const int hf = lane >> 4;
  const int tile_m = blockIdx.y * 128 + wm * 32;
  const int tile_n = blockIdx.x * 128 + wn * 64;

  v8f acc[2][4] = {};
  for (int kk = 0; kk < K; kk += 32) {
    v16bf a[2], bfr[4];
#pragma unroll
    for (int mi = 0; mi < 2; ++mi) {
      const __bf16* p = A + (size_t)(tile_m + mi * 16 + lm) * K + kk + hf * 8;
      a[mi] = load_frag(p);
      __builtin_prefetch(p + 32, 0, 1);   // global_prefetch_b8 next K-step
    }
#pragma unroll
    for (int ni = 0; ni < 4; ++ni) {
      int col = tile_n + ni * 16 + lm;
      col = (col < N) ? col : (N - 1);    // clamp edge loads (logits N=50257)
      const __bf16* p = BT + (size_t)col * K + kk + hf * 8;
      bfr[ni] = load_frag(p);
    }
#pragma unroll
    for (int mi = 0; mi < 2; ++mi)
#pragma unroll
      for (int ni = 0; ni < 4; ++ni)
        acc[mi][ni] = __builtin_amdgcn_wmma_f32_16x16x32_bf16(
            false, a[mi], false, bfr[ni], (short)0, acc[mi][ni], false, false);
  }

  float* outf = (float*)outv;
  __bf16* outb = (__bf16*)outv;
#pragma unroll
  for (int mi = 0; mi < 2; ++mi)
#pragma unroll
    for (int ni = 0; ni < 4; ++ni) {
      int col = tile_n + ni * 16 + lm;
      if (col < N) {
        float bv = bias ? bias[col] : 0.f;
#pragma unroll
        for (int r = 0; r < 8; ++r) {
          int row = tile_m + mi * 16 + r + 8 * hf;   // D layout: m = r + 8*half
          float v = acc[mi][ni][r] + bv;
          if (resid) v += resid[(size_t)row * N + col];
          if (GELU) v = gelu_f(v);
          if (OUTBF16) outb[(size_t)row * N + col] = f2bf(v);
          else         outf[(size_t)row * N + col] = v;
        }
      }
    }
}

// ---- Flash attention, one wave per 16-query tile -------------------------
// Trick: compute S^T = K @ Q^T so the score-tile D layout (lane = q column,
// regs = k positions) is exactly the A-fragment layout required by P @ V.
__global__ __launch_bounds__(128) void attn_kernel(const __bf16* __restrict__ Qb,
                                                   const __bf16* __restrict__ Kb,
                                                   const __bf16* __restrict__ VT,
                                                   __bf16* __restrict__ O) {
  const int lane = threadIdx.x & 31;
  const int wave = threadIdx.x >> 5;
  const int lm = lane & 15;
  const int hf = lane >> 4;
  const int bh = blockIdx.x;                 // b*H + h
  const int b = bh / Hq, h = bh % Hq;
  const int q0 = blockIdx.y * 64 + wave * 16;
  const size_t base = (size_t)bh * Sq * 64;

  v16bf qf[2];                               // Q as B-matrix [dk, q] frags
#pragma unroll
  for (int ks = 0; ks < 2; ++ks)
    qf[ks] = load_frag(Qb + base + (size_t)(q0 + lm) * 64 + ks * 32 + hf * 8);

  v8f oa[4] = {};                            // O[q, dk] accumulators (4 x 16 dk)
  float m_run = -1.0e30f, l_run = 0.f;
  const int qcol = q0 + lm;                  // this lane's q column in S^T
  const int nk = (q0 + 47) >> 5;             // ceil((q0+16)/32) k-blocks (causal)

  for (int it = 0; it < nk; ++it) {
    int k0 = it * 32;
    v8f st[2] = {};                          // S^T tiles: rows k0+[0,16)+16*mi, cols q
#pragma unroll
    for (int mi = 0; mi < 2; ++mi)
#pragma unroll
      for (int ks = 0; ks < 2; ++ks) {
        v16bf kf = load_frag(Kb + base + (size_t)(k0 + mi * 16 + lm) * 64 + ks * 32 + hf * 8);
        st[mi] = __builtin_amdgcn_wmma_f32_16x16x32_bf16(
            false, kf, false, qf[ks], (short)0, st[mi], false, false);
      }

    float sv[2][8];
    float tmax = -1.0e30f;
#pragma unroll
    for (int mi = 0; mi < 2; ++mi)
#pragma unroll
      for (int r = 0; r < 8; ++r) {
        int kidx = k0 + mi * 16 + 8 * hf + r;
        float v = st[mi][r];
        v = (kidx <= qcol) ? v : -1.0e30f;   // causal mask
        sv[mi][r] = v;
        tmax = fmaxf(tmax, v);
      }
    tmax = fmaxf(tmax, __shfl_xor(tmax, 16, 32));   // other 16 k's of this q
    float m_new = fmaxf(m_run, tmax);
    float alpha = __expf(m_run - m_new);
    float tsum = 0.f;
    v16bf pf = {};                           // P fragment (A-matrix [q, k])
#pragma unroll
    for (int mi = 0; mi < 2; ++mi)
#pragma unroll
      for (int r = 0; r < 8; ++r) {
        float p = __expf(sv[mi][r] - m_new);
        tsum += p;
        pf[mi * 8 + r] = f2bf(p);            // elem j -> k = 16*(j>>3)+8*hf+(j&7)
      }
    tsum += __shfl_xor(tsum, 16, 32);
    l_run = l_run * alpha + tsum;
    m_run = m_new;

    float ar[8];                             // alpha per O-row (q = r + 8*hf)
#pragma unroll
    for (int r = 0; r < 8; ++r) ar[r] = __shfl(alpha, r + 8 * hf, 32);
#pragma unroll
    for (int d4 = 0; d4 < 4; ++d4) {
      v16bf vf = load_frag(VT + base + (size_t)(d4 * 16 + lm) * Sq + k0 + hf * 8);
#pragma unroll
      for (int r = 0; r < 8; ++r) oa[d4][r] *= ar[r];
      oa[d4] = __builtin_amdgcn_wmma_f32_16x16x32_bf16(
          false, pf, false, vf, (short)0, oa[d4], false, false);
    }
  }

  float rc = 1.0f / l_run;
  float lr[8];
#pragma unroll
  for (int r = 0; r < 8; ++r) lr[r] = __shfl(rc, r + 8 * hf, 32);
#pragma unroll
  for (int d4 = 0; d4 < 4; ++d4)
#pragma unroll
    for (int r = 0; r < 8; ++r) {
      int srow = q0 + r + 8 * hf;
      int col = h * 64 + d4 * 16 + lm;
      O[((size_t)b * Sq + srow) * Dq + col] = f2bf(oa[d4][r] * lr[r]);
    }
}

// ---- orchestration -------------------------------------------------------

extern "C" void kernel_launch(void* const* d_in, const int* in_sizes, int n_in,
                              void* d_out, int out_size, void* d_ws, size_t ws_size,
                              hipStream_t stream) {
  (void)in_sizes; (void)n_in; (void)out_size; (void)ws_size;
  const int*   ids    = (const int*)  d_in[0];
  const float* wte    = (const float*)d_in[1];
  const float* wpe    = (const float*)d_in[2];
  const float* ln1_s  = (const float*)d_in[3];
  const float* ln1_b  = (const float*)d_in[4];
  const float* attn_w = (const float*)d_in[5];
  const float* attn_b = (const float*)d_in[6];
  const float* proj_w = (const float*)d_in[7];
  const float* proj_b = (const float*)d_in[8];
  const float* ln2_s  = (const float*)d_in[9];
  const float* ln2_b  = (const float*)d_in[10];
  const float* mlp_w1 = (const float*)d_in[11];
  const float* mlp_b1 = (const float*)d_in[12];
  const float* mlp_w2 = (const float*)d_in[13];
  const float* mlp_b2 = (const float*)d_in[14];
  const float* lnf_s  = (const float*)d_in[15];
  const float* lnf_b  = (const float*)d_in[16];
  float* logits = (float*)d_out;

  char* ws = (char*)d_ws;
  size_t off = 0;
  auto carve = [&](size_t bytes) -> char* {
    off = (off + 255) & ~(size_t)255;
    char* p = ws + off;
    off += bytes;
    return p;
  };
  __bf16* wteb = (__bf16*)carve((size_t)Vq * Dq * 2);       // 77 MB
  __bf16* wT   = (__bf16*)carve((size_t)Dq * Fq * 2);       // per-layer weight slab
  float*  x    = (float*) carve((size_t)Mq * Dq * 4);       // residual stream
  __bf16* hbf  = (__bf16*)carve((size_t)Mq * Dq * 2);       // LN output
  float*  qkv  = (float*) carve((size_t)Mq * 3 * Dq * 4);
  __bf16* Qb   = (__bf16*)carve((size_t)Mq * Dq * 2);
  __bf16* Kb   = (__bf16*)carve((size_t)Mq * Dq * 2);
  __bf16* VT   = (__bf16*)carve((size_t)Mq * Dq * 2);
  __bf16* obf  = (__bf16*)carve((size_t)Mq * Dq * 2);
  __bf16* mbf  = (__bf16*)carve((size_t)Mq * Fq * 2);

  auto cdiv = [](long long a, long long b) { return (int)((a + b - 1) / b); };

  cvt_bf16_kernel<<<cdiv((long long)Vq * Dq, 256), 256, 0, stream>>>(wte, wteb, (long long)Vq * Dq);
  embed_kernel<<<cdiv(Mq * Dq, 256), 256, 0, stream>>>(ids, wte, wpe, x);

  for (int l = 0; l < Lq; ++l) {
    // attention
    ln_kernel<<<Mq, 256, 0, stream>>>(x, ln1_s + l * Dq, ln1_b + l * Dq, hbf);
    transpose_bf16_kernel<<<cdiv((long long)Dq * 3 * Dq, 256), 256, 0, stream>>>(
        attn_w + (size_t)l * Dq * 3 * Dq, wT, Dq, 3 * Dq);
    gemm_kernel<false, false><<<dim3(cdiv(3 * Dq, 128), Mq / 128), 256, 0, stream>>>(
        hbf, wT, attn_b + l * 3 * Dq, nullptr, qkv, 3 * Dq, Dq);
    qkv_split_kernel<<<cdiv(Mq * Dq, 256), 256, 0, stream>>>(qkv, Qb, Kb, VT);
    attn_kernel<<<dim3(Bq * Hq, Sq / 64), 128, 0, stream>>>(Qb, Kb, VT, obf);
    transpose_bf16_kernel<<<cdiv((long long)Dq * Dq, 256), 256, 0, stream>>>(
        proj_w + (size_t)l * Dq * Dq, wT, Dq, Dq);
    gemm_kernel<false, false><<<dim3(cdiv(Dq, 128), Mq / 128), 256, 0, stream>>>(
        obf, wT, proj_b + l * Dq, x, x, Dq, Dq);
    // MLP
    ln_kernel<<<Mq, 256, 0, stream>>>(x, ln2_s + l * Dq, ln2_b + l * Dq, hbf);
    transpose_bf16_kernel<<<cdiv((long long)Dq * Fq, 256), 256, 0, stream>>>(
        mlp_w1 + (size_t)l * Dq * Fq, wT, Dq, Fq);
    gemm_kernel<true, true><<<dim3(cdiv(Fq, 128), Mq / 128), 256, 0, stream>>>(
        hbf, wT, mlp_b1 + l * Fq, nullptr, mbf, Fq, Dq);
    transpose_bf16_kernel<<<cdiv((long long)Fq * Dq, 256), 256, 0, stream>>>(
        mlp_w2 + (size_t)l * Fq * Dq, wT, Fq, Dq);
    gemm_kernel<false, false><<<dim3(cdiv(Dq, 128), Mq / 128), 256, 0, stream>>>(
        mbf, wT, mlp_b2 + l * Dq, x, x, Dq, Fq);
  }

  // final LN + tied LM head (wte already [V, D] = [N, K])
  ln_kernel<<<Mq, 256, 0, stream>>>(x, lnf_s, lnf_b, hbf);
  gemm_kernel<false, false><<<dim3(cdiv(Vq, 128), Mq / 128), 256, 0, stream>>>(
      hbf, wteb, nullptr, nullptr, logits, Vq, Dq);
}
